// SpikingLinear_49306224558691
// MI455X (gfx1250) — compile-verified, matrix operand order checked by reading, not run on hardware
//
#include <hip/hip_runtime.h>

#define S_SEG 10
#define TEMP_C 50.0f

typedef __bf16 bf16;
typedef __attribute__((ext_vector_type(16))) __bf16 v16bf;
typedef __attribute__((ext_vector_type(8)))  __bf16 v8bf;
typedef __attribute__((ext_vector_type(4)))  __bf16 v4bf;
typedef __attribute__((ext_vector_type(8)))  float  v8f;
typedef __attribute__((ext_vector_type(2)))  float  v2f;
typedef __attribute__((ext_vector_type(2)))  __bf16 v2bf;

#if defined(__has_builtin)
#  if __has_builtin(__builtin_amdgcn_sched_group_barrier)
#    define SCHED_GROUP(mask, cnt, id) __builtin_amdgcn_sched_group_barrier((mask), (cnt), (id))
#  else
#    define SCHED_GROUP(mask, cnt, id)
#  endif
#else
#  define SCHED_GROUP(mask, cnt, id)
#endif

constexpr int BS   = 2048;
constexpr int DIN  = 1024;
constexpr int DOUT = 1024;
constexpr int NCOL = DOUT * S_SEG * 3;   // 30720 columns of param_W
constexpr int BM   = 32;                 // batch rows per workgroup
constexpr int KT   = 32;                 // K per WMMA step
constexpr int NTR  = 31;                 // real n-tiles: 30 param (10 s * 3) + 1 norm
constexpr int KPA  = 40;                 // padded K stride (halves) for A rows in LDS
constexpr int KPB  = 40;                 // padded K stride (halves) for B columns in LDS
constexpr int NPAD = 497;                // padded row stride (floats) for C scratch
constexpr int STAGE_A = BM * KPA;        // halves
constexpr int SMEM_BYTES = BM * NPAD * 4; // 63616 B; stage region (2560+40960 B) aliases it

// ---------------- fp32 -> bf16 conversion (runs once per launch) ----------------
__global__ void cvt_f32_bf16(const float* __restrict__ src, bf16* __restrict__ dst, int n2) {
  int i = blockIdx.x * blockDim.x + threadIdx.x;
  const int stride = gridDim.x * blockDim.x;
  for (; i < n2; i += stride) {
    v2f v = *(const v2f*)(src + 2 * (size_t)i);
    v2bf o;
    o.x = (bf16)v.x;
    o.y = (bf16)v.y;
    *(v2bf*)(dst + 2 * (size_t)i) = o;
  }
}

// ---------------- fused GEMM (bf16 WMMA) + spiking epilogue ----------------
__global__ __launch_bounds__(256)
void fused_spiking_kernel(const bf16* __restrict__ xbf,
                          const bf16* __restrict__ wbf,   // param_W bf16, row-major [DIN][NCOL]
                          const bf16* __restrict__ nbf,   // norm_W  bf16, row-major [DIN][DOUT]
                          const float* __restrict__ t,
                          const float* __restrict__ param_b,
                          const float* __restrict__ norm_b,
                          float* __restrict__ out) {
  __shared__ __align__(16) unsigned char smem[SMEM_BYTES];
  bf16*  As = (bf16*)smem;                // [BM][KPA]
  bf16*  Bs = (bf16*)smem + STAGE_A;      // [32*16][KPB]  (n-major, k contiguous)
  float* Cs = (float*)smem;               // [BM][NPAD]    (aliases stage, used after k-loop)

  const int tid  = threadIdx.x;
  const int lane = tid & 31;
  const int wave = tid >> 5;
  const int mt   = wave & 1;              // m-tile 0..1
  const int tn0  = wave >> 1;             // base n-tile 0..3 (stride 4) -> covers 0..31

  const int j0 = blockIdx.x * 16;         // dout tile
  const int b0 = blockIdx.y * BM;         // batch tile

  v8f acc[8] = {};

  const int arow = tid >> 3;
  const int ac4  = (tid & 7) * 4;
  const int frow = lane & 15;
  const int kg   = lane >> 4;

  // ---- B-staging ownership: fixed per thread, branch resolved once ----
  // tn = tid>>3 (0..31), q = tid&3 (column quad), kbase = (tid>>2)&1; k = kbase + 2*it
  // tiles 30 and 31 both map to the norm panel; tile 31 is compute padding.
  const int stn = tid >> 3;
  const int sq  = tid & 3;
  const int skb = (tid >> 2) & 1;
  size_t rowstride;
  const bf16* srcp;
  if (stn < 30) {
    const int s  = stn / 3;
    const int cg = stn - s * 3;
    srcp = wbf + s * (DOUT * 3) + j0 * 3 + cg * 16 + sq * 4;
    rowstride = NCOL;
  } else {
    srcp = nbf + j0 + sq * 4;
    rowstride = DOUT;
  }
  srcp += (size_t)skb * rowstride;
  bf16* const dstd = Bs + (stn * 16 + sq * 4) * KPB + skb;

  for (int k0 = 0; k0 < DIN; k0 += KT) {
    __syncthreads();
    // ---- stage A: 32x32 bf16, 8B vector copies ----
    {
      const unsigned long long v =
          *(const unsigned long long*)(xbf + (size_t)(b0 + arow) * DIN + k0 + ac4);
      *(unsigned long long*)(As + arow * KPA + ac4) = v;
    }
    // ---- stage B: 16 batched global b64 loads, then 64 ds_store_b16 (transpose) ----
    {
      v4bf vals[16];
      #pragma unroll
      for (int it = 0; it < 16; ++it)
        vals[it] = *(const v4bf*)(srcp + (size_t)(2 * it) * rowstride);
      if (k0 + KT < DIN) {
        __builtin_prefetch((const void*)(srcp + (size_t)(KT +  0) * rowstride), 0, 3);
        __builtin_prefetch((const void*)(srcp + (size_t)(KT +  8) * rowstride), 0, 3);
        __builtin_prefetch((const void*)(srcp + (size_t)(KT + 16) * rowstride), 0, 3);
        __builtin_prefetch((const void*)(srcp + (size_t)(KT + 24) * rowstride), 0, 3);
      }
      #pragma unroll
      for (int it = 0; it < 16; ++it) {
        bf16* dp = dstd + 2 * it;
        dp[0 * KPB] = vals[it][0];
        dp[1 * KPB] = vals[it][1];
        dp[2 * KPB] = vals[it][2];
        dp[3 * KPB] = vals[it][3];
      }
      srcp += (size_t)KT * rowstride;
      SCHED_GROUP(0x020, 16, 1);   // 16 VMEM reads first (staggered loadcnt waits)
      SCHED_GROUP(0x200, 64, 1);   // then the DS write burst
    }
    __syncthreads();

    // ---- A fragment: lane l -> row (l&15), k-group (l>>4); ISA 16-bit A layout ----
    const bf16* ap = As + (mt * 16 + frow) * KPA + kg * 8;
    const v8bf a0 = *(const v8bf*)ap;
    const v8bf a1 = *(const v8bf*)(ap + 16);
    v16bf af;
    #pragma unroll
    for (int e = 0; e < 8; ++e) { af[e] = a0[e]; af[e + 8] = a1[e]; }

    // ---- issue all 16 B-fragment DS reads, then 8 back-to-back WMMAs ----
    v8bf bl[8], bh[8];
    #pragma unroll
    for (int u = 0; u < 8; ++u) {
      const int tn = tn0 + 4 * u;
      const bf16* bp = Bs + (tn * 16 + frow) * KPB + kg * 8;
      bl[u] = *(const v8bf*)bp;
      bh[u] = *(const v8bf*)(bp + 16);
    }
    #pragma unroll
    for (int u = 0; u < 8; ++u) {
      v16bf bfrag;
      #pragma unroll
      for (int e = 0; e < 8; ++e) { bfrag[e] = bl[u][e]; bfrag[e + 8] = bh[u][e]; }
      acc[u] = __builtin_amdgcn_wmma_f32_16x16x32_bf16(
          false, af, false, bfrag, (short)0, acc[u], false, false);
    }
    SCHED_GROUP(0x100, 18, 0);   // DS read group (A pair + 8 B pairs)
    SCHED_GROUP(0x008, 8, 0);    // WMMA group -> staggered s_wait_dscnt
  }

  __syncthreads();
  // ---- spill accumulators (16x16 f32 C layout: VGPR v / lane group -> M, lane&15 -> N) ----
  #pragma unroll
  for (int i = 0; i < 8; ++i) {
    const int tn = tn0 + 4 * i;
    if (tn < NTR) {                        // drop the padding tile (outside WMMA: safe)
      const int n = tn * 16 + frow;
      const int m = mt * 16 + kg * 8;
      #pragma unroll
      for (int v = 0; v < 8; ++v)
        Cs[(size_t)(m + v) * NPAD + n] = acc[i][v];
    }
  }
  __syncthreads();

  // ---- spiking epilogue: one (batch,dout) pair per iteration ----
  for (int p = tid; p < BM * 16; p += 256) {
    const int row = p >> 4;
    const int jl  = p & 15;
    const int b   = b0 + row;
    const int j   = j0 + jl;
    const float* crow = Cs + (size_t)row * NPAD;
    const int o0 = 3 * jl;

    float logits[S_SEG], slp[S_SEG], icp[S_SEG];
    #pragma unroll
    for (int s = 0; s < S_SEG; ++s) {
      logits[s] = crow[s * 48 + o0 + 0] + param_b[s * 3072 + 3 * j + 0];
      slp[s]    = tanhf(crow[s * 48 + o0 + 1] + param_b[s * 3072 + 3 * j + 1]);
      icp[s]    = crow[s * 48 + o0 + 2] + param_b[s * 3072 + 3 * j + 2];
    }
    const float nv = crow[480 + jl] + norm_b[j];
    const float tt = t[b];

    // softmax over segments
    float mx = logits[0];
    #pragma unroll
    for (int s = 1; s < S_SEG; ++s) mx = fmaxf(mx, logits[s]);
    float w[S_SEG], sum = 0.f;
    #pragma unroll
    for (int s = 0; s < S_SEG; ++s) { w[s] = __expf(logits[s] - mx); sum += w[s]; }
    const float inv = 1.f / sum;

    // keypoints = 2*cumsum - 1, kp[0] = -1
    float kp[S_SEG + 1];
    kp[0] = -1.f;
    float cum = 0.f;
    #pragma unroll
    for (int s = 0; s < S_SEG; ++s) { cum += w[s] * inv; kp[s + 1] = 2.f * cum - 1.f; }

    float vf[S_SEG], rs[S_SEG];
    float lm = 0.f, rm = 0.f, integ = 0.f;
    #pragma unroll
    for (int s = 0; s < S_SEG; ++s) {
      const float st = kp[s], en = kp[s + 1];
      rs[s] = slp[s] * tt + icp[s];
      const bool valid = ((tt >= st) && (tt < en)) || ((tt == 1.f) && (en == 1.f));
      vf[s] = valid ? 1.f : 0.f;
      lm += (tt - st) * vf[s];
      rm += (en - tt) * vf[s];
      integ += 0.5f * slp[s] * (en * en - st * st) + icp[s] * (w[s] * inv);
    }
    float sumL = 0.f, sumV = 0.f, sumR = 0.f;
    #pragma unroll
    for (int s = 0; s < S_SEG; ++s) {
      const float lv = vf[(s == 0) ? 0 : s - 1];
      const float rv = vf[(s == S_SEG - 1) ? (S_SEG - 1) : s + 1];
      sumL += rs[s] * lv;
      sumV += rs[s] * vf[s];
      sumR += rs[s] * rv;
    }
    const float lw = 1.f / (1.f + __expf(TEMP_C * lm));   // sigmoid(-T*lm)
    const float rw = 1.f / (1.f + __expf(TEMP_C * rm));
    const float sw = 1.f - lw - rw;
    const float result = lw * sumL + sw * sumV + rw * sumR;
    out[(size_t)b * DOUT + j] = result - 0.5f * integ + nv;
  }
}

extern "C" void kernel_launch(void* const* d_in, const int* in_sizes, int n_in,
                              void* d_out, int out_size, void* d_ws, size_t ws_size,
                              hipStream_t stream) {
  const float* x  = (const float*)d_in[0];
  const float* t  = (const float*)d_in[1];
  const float* pW = (const float*)d_in[2];
  const float* pb = (const float*)d_in[3];
  const float* nW = (const float*)d_in[4];
  const float* nb = (const float*)d_in[5];
  float* out = (float*)d_out;

  // workspace layout: param_W bf16 | x bf16 | norm_W bf16  (~66 MB total)
  bf16* wbf = (bf16*)d_ws;
  bf16* xbf = wbf + (size_t)DIN * NCOL;
  bf16* nbf = xbf + (size_t)BS * DIN;

  cvt_f32_bf16<<<2048, 256, 0, stream>>>(pW, wbf, (DIN * NCOL) / 2);
  cvt_f32_bf16<<<512,  256, 0, stream>>>(x,  xbf, (BS * DIN) / 2);
  cvt_f32_bf16<<<512,  256, 0, stream>>>(nW, nbf, (DIN * DOUT) / 2);

  dim3 grid(DOUT / 16, BS / BM);   // 64 x 64 workgroups
  fused_spiking_kernel<<<grid, 256, 0, stream>>>(xbf, wbf, nbf, t, pb, nb, out);
}
